// DrugRNACrossAttention_22093311771377
// MI455X (gfx1250) — matrix operand order repaired and checked by reading, not compiled, not run on hardware
//
#include <hip/hip_runtime.h>
#include <math.h>

typedef __bf16 bf16_t;
typedef __attribute__((ext_vector_type(16))) __bf16 v16bf;
typedef __attribute__((ext_vector_type(8)))  __bf16 v8bf;
typedef __attribute__((ext_vector_type(8)))  float  v8f;

#define HEADS 8
#define DH    32
#define HD    256   // HEADS*DH
#define WPB   4     // waves per block (128 threads)

// ---------------- elementwise fp32 -> bf16 ----------------
__global__ void cvt_bf16_kernel(const float* __restrict__ src, bf16_t* __restrict__ dst, int n) {
    int i = blockIdx.x * blockDim.x + threadIdx.x;
    if (i < n) dst[i] = (bf16_t)src[i];
}

// ---------------- transpose fp32 [R,C] -> bf16 [C,R] ----------------
__global__ void transpose_bf16_kernel(const float* __restrict__ src, bf16_t* __restrict__ dst,
                                      int R, int C) {
    int i = blockIdx.x * blockDim.x + threadIdx.x;
    if (i < R * C) {
        int r = i / C, c = i - r * C;
        dst[c * R + r] = (bf16_t)src[i];
    }
}

// ---------------- projection GEMM: C = X[MxD] * W[DxN], WT = W^T [NxD] bf16 ----------------
// transOut==0: OUT bf16 [M x N] row-major.  transOut==1: OUT bf16 [N x M] (for V^T).
__global__ __launch_bounds__(128)
void proj_gemm_kernel(const bf16_t* __restrict__ X, const bf16_t* __restrict__ WT,
                      bf16_t* __restrict__ OUT, int M, int D, int N, int transOut) {
    const int lane   = threadIdx.x & 31;
    const int lane16 = lane & 15;
    const int half   = lane >> 4;
    const int wave   = blockIdx.x * WPB + (threadIdx.x >> 5);
    const int ntiles = N >> 4;
    const int m0 = (wave / ntiles) << 4;
    const int n0 = (wave % ntiles) << 4;
    if (m0 >= M) return;

    const bf16_t* xrow = X  + (size_t)(m0 + lane16) * D;
    const bf16_t* wrow = WT + (size_t)(n0 + lane16) * D;
    v8f c = {};
    for (int k0 = 0; k0 < D; k0 += 32) {
        v8bf alo = *(const v8bf*)(xrow + k0 + 16 * half);
        v8bf ahi = *(const v8bf*)(xrow + k0 + 16 * half + 8);
        v16bf a;
#pragma unroll
        for (int i = 0; i < 8; ++i) { a[i] = alo[i]; a[8 + i] = ahi[i]; }
        v16bf b = *(const v16bf*)(wrow + k0 + 16 * half);
        c = __builtin_amdgcn_wmma_f32_16x16x32_bf16(false, a, false, b, (short)0, c, false, false);
    }
    if (!transOut) {
#pragma unroll
        for (int v = 0; v < 8; ++v)
            OUT[(size_t)(m0 + v + 8 * half) * N + n0 + lane16] = (bf16_t)c[v];
    } else {
        v8bf pk;
#pragma unroll
        for (int v = 0; v < 8; ++v) pk[v] = (bf16_t)c[v];
        *(v8bf*)(OUT + (size_t)(n0 + lane16) * M + m0 + 8 * half) = pk;
    }
}

// ---------------- flash attention: one wave per (16-query tile, head) ----------------
// Q [Nq x 256] bf16, K [Nk x 256] bf16, VT [256 x Nk] bf16, CTX [Nq x 256] bf16.
__global__ __launch_bounds__(128)
void flash_attn_kernel(const bf16_t* __restrict__ Q, const bf16_t* __restrict__ K,
                       const bf16_t* __restrict__ VT, const int* __restrict__ qb,
                       const int* __restrict__ kb, bf16_t* __restrict__ CTX,
                       int Nq, int Nk, float scale) {
    __shared__ __align__(16) bf16_t pbuf[WPB][16][32];
    __shared__ int qbs[WPB][16];

    const int lane   = threadIdx.x & 31;
    const int lane16 = lane & 15;
    const int half   = lane >> 4;
    const int w      = threadIdx.x >> 5;
    const int wave   = blockIdx.x * WPB + w;
    const int head   = wave & (HEADS - 1);
    const int q0     = (wave / HEADS) << 4;
    if (q0 >= Nq) return;

    if (lane < 16) qbs[w][lane] = qb[q0 + lane];

    // sorted batch indices -> contiguous valid key range for this query tile
    const int qbmin = qb[q0];
    const int qbmax = qb[q0 + 15];
    int lo = 0, h1 = Nk;
    while (lo < h1) { int mid = (lo + h1) >> 1; if (kb[mid] <  qbmin) lo = mid + 1; else h1 = mid; }
    int hi = lo, h2 = Nk;
    while (hi < h2) { int mid = (hi + h2) >> 1; if (kb[mid] <= qbmax) hi = mid + 1; else h2 = mid; }
    const int k0start = lo & ~31;

    // Q fragment (A layout, 16x32 bf16), loaded once
    const bf16_t* qrow = Q + (size_t)(q0 + lane16) * HD + head * DH;
    v8bf qlo = *(const v8bf*)(qrow + 16 * half);
    v8bf qhi = *(const v8bf*)(qrow + 16 * half + 8);
    v16bf aq;
#pragma unroll
    for (int i = 0; i < 8; ++i) { aq[i] = qlo[i]; aq[8 + i] = qhi[i]; }

    float m[8], lsum[8];
    v8f o0 = {}, o1 = {};
#pragma unroll
    for (int v = 0; v < 8; ++v) { m[v] = -1e30f; lsum[v] = 0.f; }

    for (int key0 = k0start; key0 < hi; key0 += 32) {
        // scores S = Q * K^T : two 16x16 tiles over this 32-key block
        v16bf bk0 = *(const v16bf*)(K + (size_t)(key0 + lane16)      * HD + head * DH + 16 * half);
        v16bf bk1 = *(const v16bf*)(K + (size_t)(key0 + 16 + lane16) * HD + head * DH + 16 * half);
        v8f z = {};
        v8f c0 = __builtin_amdgcn_wmma_f32_16x16x32_bf16(false, aq, false, bk0, (short)0, z, false, false);
        v8f c1 = __builtin_amdgcn_wmma_f32_16x16x32_bf16(false, aq, false, bk1, (short)0, z, false, false);

        const int kb0 = kb[key0 + lane16];
        const int kb1 = kb[key0 + 16 + lane16];

#pragma unroll
        for (int v = 0; v < 8; ++v) {
            const int row = v + 8 * half;           // score row handled by this half-wave
            const int qbv = qbs[w][row];
            float s0 = (kb0 == qbv) ? c0[v] * scale : -1e30f;
            float s1 = (kb1 == qbv) ? c1[v] * scale : -1e30f;
            // row max over the 16 lanes of this half
            float bm = fmaxf(s0, s1);
            bm = fmaxf(bm, __shfl_xor(bm, 1));
            bm = fmaxf(bm, __shfl_xor(bm, 2));
            bm = fmaxf(bm, __shfl_xor(bm, 4));
            bm = fmaxf(bm, __shfl_xor(bm, 8));
            const float mnew  = fmaxf(m[v], bm);
            const float alpha = __expf(m[v] - mnew);
            const float p0 = __expf(s0 - mnew);
            const float p1 = __expf(s1 - mnew);
            float bs = p0 + p1;
            bs += __shfl_xor(bs, 1);
            bs += __shfl_xor(bs, 2);
            bs += __shfl_xor(bs, 4);
            bs += __shfl_xor(bs, 8);
            lsum[v] = lsum[v] * alpha + bs;
            m[v] = mnew;
            o0[v] *= alpha;
            o1[v] *= alpha;
            pbuf[w][row][lane16]      = (bf16_t)p0;   // C-layout -> row-major staging
            pbuf[w][row][16 + lane16] = (bf16_t)p1;
        }

        // reload P in A layout (wave-private LDS; DS ops in-order within wave)
        v8bf plo = *(const v8bf*)(&pbuf[w][lane16][16 * half]);
        v8bf phi = *(const v8bf*)(&pbuf[w][lane16][16 * half + 8]);
        v16bf ap;
#pragma unroll
        for (int i = 0; i < 8; ++i) { ap[i] = plo[i]; ap[8 + i] = phi[i]; }

        // V fragments (B layout) straight from V^T [256 x Nk]
        v16bf bv0 = *(const v16bf*)(VT + (size_t)(head * DH + lane16)      * Nk + key0 + 16 * half);
        v16bf bv1 = *(const v16bf*)(VT + (size_t)(head * DH + 16 + lane16) * Nk + key0 + 16 * half);
        o0 = __builtin_amdgcn_wmma_f32_16x16x32_bf16(false, ap, false, bv0, (short)0, o0, false, false);
        o1 = __builtin_amdgcn_wmma_f32_16x16x32_bf16(false, ap, false, bv1, (short)0, o1, false, false);
    }

#pragma unroll
    for (int v = 0; v < 8; ++v) {
        const int row = q0 + v + 8 * half;
        const float inv = lsum[v] > 0.f ? 1.0f / lsum[v] : 0.f;
        CTX[(size_t)row * HD + head * DH + lane16]      = (bf16_t)(o0[v] * inv);
        CTX[(size_t)row * HD + head * DH + 16 + lane16] = (bf16_t)(o1[v] * inv);
    }
}

// ---------------- output GEMM: OUT f32 [M x N] = CTX[M x D] * Wo[D x N] + bias ----------------
__global__ __launch_bounds__(128)
void out_gemm_kernel(const bf16_t* __restrict__ CTX, const bf16_t* __restrict__ WoT,
                     const float* __restrict__ bias, float* __restrict__ OUT,
                     int M, int D, int N) {
    const int lane   = threadIdx.x & 31;
    const int lane16 = lane & 15;
    const int half   = lane >> 4;
    const int wave   = blockIdx.x * WPB + (threadIdx.x >> 5);
    const int ntiles = N >> 4;
    const int m0 = (wave / ntiles) << 4;
    const int n0 = (wave % ntiles) << 4;
    if (m0 >= M) return;

    const bf16_t* xrow = CTX + (size_t)(m0 + lane16) * D;
    const bf16_t* wrow = WoT + (size_t)(n0 + lane16) * D;
    v8f c = {};
    for (int k0 = 0; k0 < D; k0 += 32) {
        v8bf alo = *(const v8bf*)(xrow + k0 + 16 * half);
        v8bf ahi = *(const v8bf*)(xrow + k0 + 16 * half + 8);
        v16bf a;
#pragma unroll
        for (int i = 0; i < 8; ++i) { a[i] = alo[i]; a[8 + i] = ahi[i]; }
        v16bf b = *(const v16bf*)(wrow + k0 + 16 * half);
        c = __builtin_amdgcn_wmma_f32_16x16x32_bf16(false, a, false, b, (short)0, c, false, false);
    }
    const float bv = bias[n0 + lane16];
#pragma unroll
    for (int v = 0; v < 8; ++v)
        OUT[(size_t)(m0 + v + 8 * half) * N + n0 + lane16] = c[v] + bv;
}

extern "C" void kernel_launch(void* const* d_in, const int* in_sizes, int n_in,
                              void* d_out, int out_size, void* d_ws, size_t ws_size,
                              hipStream_t stream) {
    (void)n_in; (void)out_size; (void)ws_size;
    const float* drug = (const float*)d_in[0];
    const float* rna  = (const float*)d_in[1];
    const float* Wq_d = (const float*)d_in[2];
    const float* Wk_r = (const float*)d_in[3];
    const float* Wv_r = (const float*)d_in[4];
    const float* Wo_d = (const float*)d_in[5];
    const float* bo_d = (const float*)d_in[6];
    const float* Wq_r = (const float*)d_in[7];
    const float* Wk_d = (const float*)d_in[8];
    const float* Wv_d = (const float*)d_in[9];
    const float* Wo_r = (const float*)d_in[10];
    const float* bo_r = (const float*)d_in[11];
    const int*   dbi  = (const int*)d_in[12];
    const int*   rbi  = (const int*)d_in[13];

    const int Nd = in_sizes[12];            // 2048
    const int Nr = in_sizes[13];            // 4096
    const int Dd = in_sizes[0] / Nd;        // 128
    const int Dr = in_sizes[1] / Nr;        // 128
    const int Hd = in_sizes[2] / Dd;        // 256
    const float scale = 1.0f / sqrtf((float)(Hd / HEADS));   // 1/sqrt(32)

    // ---- workspace carve-out ----
    char* p = (char*)d_ws;
    auto alloc = [&](size_t bytes) -> char* {
        char* r = p; p += (bytes + 255) & ~(size_t)255; return r;
    };
    bf16_t* Xd   = (bf16_t*)alloc((size_t)Nd * Dd * 2);
    bf16_t* Xr   = (bf16_t*)alloc((size_t)Nr * Dr * 2);
    bf16_t* WqdT = (bf16_t*)alloc((size_t)Hd * Dd * 2);
    bf16_t* WkrT = (bf16_t*)alloc((size_t)Hd * Dr * 2);
    bf16_t* WvrT = (bf16_t*)alloc((size_t)Hd * Dr * 2);
    bf16_t* WqrT = (bf16_t*)alloc((size_t)Hd * Dr * 2);
    bf16_t* WkdT = (bf16_t*)alloc((size_t)Hd * Dd * 2);
    bf16_t* WvdT = (bf16_t*)alloc((size_t)Hd * Dd * 2);
    bf16_t* WodT = (bf16_t*)alloc((size_t)Dd * Hd * 2);
    bf16_t* WorT = (bf16_t*)alloc((size_t)Dr * Hd * 2);
    bf16_t* Qd   = (bf16_t*)alloc((size_t)Nd * Hd * 2);
    bf16_t* Kr   = (bf16_t*)alloc((size_t)Nr * Hd * 2);
    bf16_t* VrT  = (bf16_t*)alloc((size_t)Hd * Nr * 2);
    bf16_t* Qr   = (bf16_t*)alloc((size_t)Nr * Hd * 2);
    bf16_t* Kd   = (bf16_t*)alloc((size_t)Nd * Hd * 2);
    bf16_t* VdT  = (bf16_t*)alloc((size_t)Hd * Nd * 2);
    bf16_t* Ctxd = (bf16_t*)alloc((size_t)Nd * Hd * 2);
    bf16_t* Ctxr = (bf16_t*)alloc((size_t)Nr * Hd * 2);

    auto grid1 = [](int n) { return dim3((unsigned)((n + 255) / 256)); };

    // 1) bf16 copies of node features
    cvt_bf16_kernel<<<grid1(Nd * Dd), 256, 0, stream>>>(drug, Xd, Nd * Dd);
    cvt_bf16_kernel<<<grid1(Nr * Dr), 256, 0, stream>>>(rna,  Xr, Nr * Dr);
    // 2) transposed bf16 weights
    transpose_bf16_kernel<<<grid1(Dd * Hd), 256, 0, stream>>>(Wq_d, WqdT, Dd, Hd);
    transpose_bf16_kernel<<<grid1(Dr * Hd), 256, 0, stream>>>(Wk_r, WkrT, Dr, Hd);
    transpose_bf16_kernel<<<grid1(Dr * Hd), 256, 0, stream>>>(Wv_r, WvrT, Dr, Hd);
    transpose_bf16_kernel<<<grid1(Dr * Hd), 256, 0, stream>>>(Wq_r, WqrT, Dr, Hd);
    transpose_bf16_kernel<<<grid1(Dd * Hd), 256, 0, stream>>>(Wk_d, WkdT, Dd, Hd);
    transpose_bf16_kernel<<<grid1(Dd * Hd), 256, 0, stream>>>(Wv_d, WvdT, Dd, Hd);
    transpose_bf16_kernel<<<grid1(Hd * Dd), 256, 0, stream>>>(Wo_d, WodT, Hd, Dd);
    transpose_bf16_kernel<<<grid1(Hd * Dr), 256, 0, stream>>>(Wo_r, WorT, Hd, Dr);

    // 3) QKV projections (one 16x16 tile per wave)
    auto gemm_blocks = [](int M, int N) { return dim3((unsigned)(((M / 16) * (N / 16)) / WPB)); };
    proj_gemm_kernel<<<gemm_blocks(Nd, Hd), 128, 0, stream>>>(Xd, WqdT, Qd,  Nd, Dd, Hd, 0);
    proj_gemm_kernel<<<gemm_blocks(Nr, Hd), 128, 0, stream>>>(Xr, WkrT, Kr,  Nr, Dr, Hd, 0);
    proj_gemm_kernel<<<gemm_blocks(Nr, Hd), 128, 0, stream>>>(Xr, WvrT, VrT, Nr, Dr, Hd, 1);
    proj_gemm_kernel<<<gemm_blocks(Nr, Hd), 128, 0, stream>>>(Xr, WqrT, Qr,  Nr, Dr, Hd, 0);
    proj_gemm_kernel<<<gemm_blocks(Nd, Hd), 128, 0, stream>>>(Xd, WkdT, Kd,  Nd, Dd, Hd, 0);
    proj_gemm_kernel<<<gemm_blocks(Nd, Hd), 128, 0, stream>>>(Xd, WvdT, VdT, Nd, Dd, Hd, 1);

    // 4) masked flash attention, both directions
    flash_attn_kernel<<<dim3((unsigned)(((Nd / 16) * HEADS) / WPB)), 128, 0, stream>>>(
        Qd, Kr, VrT, dbi, rbi, Ctxd, Nd, Nr, scale);
    flash_attn_kernel<<<dim3((unsigned)(((Nr / 16) * HEADS) / WPB)), 128, 0, stream>>>(
        Qr, Kd, VdT, rbi, dbi, Ctxr, Nr, Nd, scale);

    // 5) output projections + bias -> f32 d_out (drug ctx, then rna ctx)
    float* out_d = (float*)d_out;
    float* out_r = out_d + (size_t)Nd * Dd;
    out_gemm_kernel<<<gemm_blocks(Nd, Dd), 128, 0, stream>>>(Ctxd, WodT, bo_d, out_d, Nd, Hd, Dd);
    out_gemm_kernel<<<gemm_blocks(Nr, Dr), 128, 0, stream>>>(Ctxr, WorT, bo_r, out_r, Nr, Hd, Dr);
}
